// RNN_23132693856478
// MI455X (gfx1250) — matrix operand order repaired and checked by reading, not compiled, not run on hardware
//
#include <hip/hip_runtime.h>

// Problem constants (from reference setup_inputs)
#define B_  1024
#define T_  512
#define I_  100
#define H_  150
#define O_  3
#define HP  160   // H padded to multiple of 32 (WMMA K / N tiling)
#define IP  128   // I padded to multiple of 32
#define HLD 168   // h_lds row stride (halfwords): 336B -> conflict-free b128 LDS reads
#define XLD 136   // x_lds row stride (halfwords): 272B -> conflict-free
#define TH_ (T_ * H_)

typedef __attribute__((ext_vector_type(16))) __bf16 v16bf;
typedef __attribute__((ext_vector_type(8)))  float  v8f;

union BFrag {
  v16bf v;
  unsigned short s[16];
  uint4 q[2];
};

// f32 -> bf16 round-to-nearest-even
__device__ __forceinline__ unsigned short f2bf(float f) {
  union { float f; unsigned u; } x; x.f = f;
  unsigned r = x.u + 0x7fffu + ((x.u >> 16) & 1u);
  return (unsigned short)(r >> 16);
}

// Build a 32x16 bf16 B-fragment (K x N) from row-major f32 weight W[n][k]
// (i.e. B = W^T), zero-padded outside [nrows, ncols].
// Layout (ISA 7.12.2): lane L -> n = n0 + (L&15); elem i<8 -> k = k0 + 8*(L>>4) + i, i>=8 -> k = k0 + 16 + 8*(L>>4) + (i-8).
__device__ __forceinline__ v16bf load_B_global(const float* __restrict__ W,
                                               int nrows, int ncols,
                                               int n0, int k0, int lane) {
  int n  = n0 + (lane & 15);
  int kb = k0 + ((lane >> 4) << 3);
  BFrag u;
#pragma unroll
  for (int i = 0; i < 16; ++i) {
    int k = kb + (i < 8 ? i : i + 8);
    float v = (n < nrows && k < ncols) ? W[n * ncols + k] : 0.0f;
    u.s[i] = f2bf(v);
  }
  return u.v;
}

// Build a 16x32 bf16 A-fragment from bf16 row-major LDS tile (stride ldk halfwords).
// Layout (ISA 7.12.2): lane L -> m = L&15, hf = L>>4;
//   elems 0..7  -> k = k0 + 8*hf + i        (16B contiguous run)
//   elems 8..15 -> k = k0 + 16 + 8*hf + i   (16B contiguous run)
__device__ __forceinline__ v16bf load_A_lds(const unsigned short* lds, int ldk,
                                            int k0, int lane) {
  int m = lane & 15, hf = lane >> 4;
  const unsigned short* base = lds + m * ldk + k0 + 8 * hf;
  BFrag u;
  u.q[0] = *(const uint4*)(base);       // ds_load_b128
  u.q[1] = *(const uint4*)(base + 16);  // ds_load_b128
  return u.v;
}

#define WMMA_BF16(a, b, c) \
  __builtin_amdgcn_wmma_f32_16x16x32_bf16(false, (a), false, (b), (short)0, (c), false, false)

// ---------------------------------------------------------------------------
// Kernel 1: persistent RNN scan.  64 WGs x 160 threads (5 waves).
// WG owns batch rows [b0, b0+16). Wave w owns H-columns [32w, 32w+32)
// as two 16-wide N-tiles sharing every A-fragment.
// Weight B-fragments live in VGPRs for all 512 steps; h broadcast via LDS.
// 4 independent WMMA chains (x/h contribution x 2 N-tiles) for ILP.
// ---------------------------------------------------------------------------
__global__ __launch_bounds__(160)
void rnn_scan_kernel(const float* __restrict__ x,  const float* __restrict__ h0,
                     const float* __restrict__ Wih, const float* __restrict__ Whh,
                     float* __restrict__ hid) {
  __shared__ unsigned short h_lds[16 * HLD];  // h_{t-1} tile, bf16, padded
  __shared__ unsigned short x_lds[16 * XLD];  // x_t tile, bf16, padded

  const int tid  = threadIdx.x;
  const int lane = tid & 31;
  const int wave = tid >> 5;          // 0..4
  const int b0   = blockIdx.x * 16;
  const int hf   = lane >> 4;
  const int nlo  = lane & 15;

  // Persistent weight fragments in VGPRs (18 frags = 144 VGPRs/lane).
  v16bf BWih[2][4];   // B = W_ih^T : K = I (pad 128), N = H (pad 160)
  v16bf BWhh[2][5];   // B = W_hh^T : K = H (pad 160), N = H (pad 160)
#pragma unroll
  for (int j = 0; j < 2; ++j) {
    int n0 = (2 * wave + j) * 16;
#pragma unroll
    for (int ks = 0; ks < 4; ++ks)
      BWih[j][ks] = load_B_global(Wih, H_, I_, n0, 32 * ks, lane);
#pragma unroll
    for (int ks = 0; ks < 5; ++ks)
      BWhh[j][ks] = load_B_global(Whh, H_, H_, n0, 32 * ks, lane);
  }

  // Init LDS: zero everything (pad cols stay zero forever), then load h0.
  for (int idx = tid; idx < 16 * XLD; idx += 160) x_lds[idx] = 0;
  for (int idx = tid; idx < 16 * HLD; idx += 160) h_lds[idx] = 0;
  __syncthreads();
  {
    // 160 threads: (row = tid&15, chunk = tid>>4) -> 15 contiguous elems
    int m = tid & 15, kc = (tid >> 4) * 15;
    const float* src = h0 + (b0 + m) * H_ + kc;
    unsigned short* dst = h_lds + m * HLD + kc;
#pragma unroll
    for (int i = 0; i < 15; ++i) dst[i] = f2bf(src[i]);
  }
  __syncthreads();

  // Staging partition: (row = tid&15, chunk = tid>>4) -> 10 contiguous elems
  const int sm = tid & 15;
  const int sk = (tid >> 4) * 10;
  const int n0a = (2 * wave) * 16 + nlo;   // N column of tile j=0
  const int n0b = n0a + 16;                // N column of tile j=1
  float* const dstA = hid + ((size_t)(b0 + 8 * hf) * T_) * H_ + n0a;
  float* const dstB = hid + ((size_t)(b0 + 8 * hf) * T_) * H_ + n0b;

  for (int t = 0; t < T_; ++t) {
    // Stage x_t tile -> LDS bf16 (only k < I; pad already zero). Prefetch t+1.
    {
      const float* src = x + ((size_t)(b0 + sm) * T_ + t) * I_ + sk;
      unsigned short* dst = x_lds + sm * XLD + sk;
#pragma unroll
      for (int i = 0; i < 10; ++i) dst[i] = f2bf(src[i]);
      if (t + 1 < T_) __builtin_prefetch(src + I_, 0, 1);  // global_prefetch_b8
    }
    __syncthreads();

    // 4 independent accumulator chains; A-fragments shared by both N-tiles.
    v8f cx0 = {}, cx1 = {}, ch0 = {}, ch1 = {};
#pragma unroll
    for (int ks = 0; ks < 4; ++ks) {
      v16bf a = load_A_lds(x_lds, XLD, 32 * ks, lane);
      cx0 = WMMA_BF16(a, BWih[0][ks], cx0);
      cx1 = WMMA_BF16(a, BWih[1][ks], cx1);
    }
#pragma unroll
    for (int ks = 0; ks < 5; ++ks) {
      v16bf a = load_A_lds(h_lds, HLD, 32 * ks, lane);
      ch0 = WMMA_BF16(a, BWhh[0][ks], ch0);
      ch1 = WMMA_BF16(a, BWhh[1][ks], ch1);
    }

    // ReLU(xw + hWhh); store hidden with guard hoisted out of the row loop.
    float keep0[8], keep1[8];
#pragma unroll
    for (int r = 0; r < 8; ++r) keep0[r] = fmaxf(cx0[r] + ch0[r], 0.0f);
#pragma unroll
    for (int r = 0; r < 8; ++r) keep1[r] = fmaxf(cx1[r] + ch1[r], 0.0f);

    if (n0a < H_) {
      float* d = dstA + (size_t)t * H_;
#pragma unroll
      for (int r = 0; r < 8; ++r) d[(size_t)r * TH_] = keep0[r];
    }
    if (n0b < H_) {
      float* d = dstB + (size_t)t * H_;
#pragma unroll
      for (int r = 0; r < 8; ++r) d[(size_t)r * TH_] = keep1[r];
    }

    __syncthreads();  // all waves done reading h_lds for this step
    // Publish h_t (bf16) for next step; padded columns get exactly 0.
    {
      unsigned short* pa = h_lds + 8 * hf * HLD + n0a;
      unsigned short* pb = h_lds + 8 * hf * HLD + n0b;
#pragma unroll
      for (int r = 0; r < 8; ++r) pa[r * HLD] = f2bf(keep0[r]);
#pragma unroll
      for (int r = 0; r < 8; ++r) pb[r * HLD] = f2bf(keep1[r]);
    }
    // next-iteration top barrier orders these writes vs. next reads
  }
}

// ---------------------------------------------------------------------------
// Kernel 2: head GEMM (logits = hidden @ W_fc^T) + hard Gumbel one-hot.
// out = y_soft + sg(y_hard - y_soft) == y_hard == onehot(argmax(logits + g)).
// One 16-row tile of flattened [B*T] per wave; 8 waves per block.
// ---------------------------------------------------------------------------
__global__ __launch_bounds__(256)
void head_kernel(const float* __restrict__ hid, const float* __restrict__ g,
                 const float* __restrict__ Wfc,
                 float* __restrict__ out, float* __restrict__ logits) {
  __shared__ float ldsD[8][16][16];
  const int tid  = threadIdx.x;
  const int lane = tid & 31;
  const int wave = tid >> 5;
  const int r0   = (blockIdx.x * 8 + wave) * 16;  // row tile base in [0, B*T)
  const int m    = lane & 15;
  const int hf   = lane >> 4;

  v16bf bW[5];
#pragma unroll
  for (int ks = 0; ks < 5; ++ks)
    bW[ks] = load_B_global(Wfc, O_, H_, 0, 32 * ks, lane);  // B = W_fc^T (N=3 of 16)

  const float* row = hid + (size_t)(r0 + m) * H_;
  v8f c = {};
#pragma unroll
  for (int ks = 0; ks < 5; ++ks) {
    int k0 = 32 * ks;
    BFrag u;
#pragma unroll
    for (int i = 0; i < 8; ++i) {
      int k = k0 + 8 * hf + i;
      u.s[i] = (k < H_) ? f2bf(row[k]) : (unsigned short)0;
    }
#pragma unroll
    for (int i = 0; i < 8; ++i) {
      int k = k0 + 16 + 8 * hf + i;
      u.s[8 + i] = (k < H_) ? f2bf(row[k]) : (unsigned short)0;
    }
    c = WMMA_BF16(u.v, bW[ks], c);
  }
  {
    int n = lane & 15;
#pragma unroll
    for (int r = 0; r < 8; ++r)
      ldsD[wave][r + 8 * hf][n] = c[r];
  }
  __syncthreads();

  if (lane < 16) {
    int rr = r0 + lane;
    float l0 = ldsD[wave][lane][0];
    float l1 = ldsD[wave][lane][1];
    float l2 = ldsD[wave][lane][2];
    logits[rr * 3 + 0] = l0;
    logits[rr * 3 + 1] = l1;
    logits[rr * 3 + 2] = l2;
    float z0 = l0 + g[rr * 3 + 0];
    float z1 = l1 + g[rr * 3 + 1];
    float z2 = l2 + g[rr * 3 + 2];
    int idx = 0; float best = z0;                 // first-max = jnp.argmax tiebreak
    if (z1 > best) { best = z1; idx = 1; }
    if (z2 > best) { best = z2; idx = 2; }
    out[rr * 3 + 0] = (idx == 0) ? 1.0f : 0.0f;
    out[rr * 3 + 1] = (idx == 1) ? 1.0f : 0.0f;
    out[rr * 3 + 2] = (idx == 2) ? 1.0f : 0.0f;
  }
}

// ---------------------------------------------------------------------------
extern "C" void kernel_launch(void* const* d_in, const int* in_sizes, int n_in,
                              void* d_out, int out_size, void* d_ws, size_t ws_size,
                              hipStream_t stream) {
  (void)in_sizes; (void)n_in; (void)out_size; (void)d_ws; (void)ws_size;
  const float* x   = (const float*)d_in[0];
  const float* h0  = (const float*)d_in[1];
  const float* g   = (const float*)d_in[2];
  const float* Wih = (const float*)d_in[3];
  const float* Whh = (const float*)d_in[4];
  const float* Wfc = (const float*)d_in[5];

  // d_out = concat(out[B,T,O], hidden[B,T,H], logits[B,T,O]) flat, f32
  float* out    = (float*)d_out;
  float* hid    = out + (size_t)B_ * T_ * O_;
  float* logits = hid + (size_t)B_ * T_ * H_;

  rnn_scan_kernel<<<B_ / 16, 160, 0, stream>>>(x, h0, Wih, Whh, hid);
  head_kernel<<<(B_ * T_) / (8 * 16), 256, 0, stream>>>(hid, g, Wfc, out, logits);
}
